// AdversarialContrastiveLoss_29394756173979
// MI455X (gfx1250) — compile-verified, matrix operand order checked by reading, not run on hardware
//
#include <hip/hip_runtime.h>
#include <hip/hip_bf16.h>

typedef __attribute__((ext_vector_type(16))) __bf16 v16bf;
typedef __attribute__((ext_vector_type(8)))  float  v8f;

#define DK      256
#define KCHUNKS 8            // DK / 32
#define MARGIN_F  0.2f
#define NEG_INF_F -1e9f
#define TILE_BYTES 8192      // 16 rows * 256 bf16 * 2B, contiguous in row-major P

union FragU { v16bf v; uint4 q[2]; };

// WMMA 16-bit fragment layout (ISA 7.12.2): lane L holds row/col M|N = L&15;
// elements 0..7 = K kb..kb+7, elements 8..15 = K kb+16..kb+23,
// kb = kc*32 + (L>=16 ? 8 : 0). A and B use the same per-lane pattern here
// because sim = P * P^T (B column j == P row j).
__device__ __forceinline__ v16bf load_frag_g(const unsigned short* __restrict__ Pb,
                                             int row, int kc, int lane) {
    const int kb = kc * 32 + ((lane >> 4) << 3);
    const unsigned short* base = Pb + (size_t)row * DK + kb;
    FragU u;
    u.q[0] = *(const uint4*)(base);
    u.q[1] = *(const uint4*)(base + 16);
    return u.v;
}

__device__ __forceinline__ v16bf load_frag_lds(const unsigned short* tile,
                                               int kc, int lane) {
    const int kb = kc * 32 + ((lane >> 4) << 3);
    const unsigned short* base = tile + (lane & 15) * DK + kb;
    FragU u;
    u.q[0] = *(const uint4*)(base);
    u.q[1] = *(const uint4*)(base + 16);
    return u.v;
}

// Async-copy one contiguous 8KB column tile (16 rows x 256 bf16) into LDS.
// 128 threads x 4 x b128 = 8KB. Tracked by ASYNCcnt (4 instructions per wave).
__device__ __forceinline__ void async_issue_tile(const unsigned short* __restrict__ Pb,
                                                 int ct, unsigned lds_base, int t) {
    unsigned voff = (unsigned)ct * TILE_BYTES + (unsigned)t * 16u;
    unsigned lds  = lds_base + (unsigned)t * 16u;
#pragma unroll
    for (int k = 0; k < 4; ++k) {
        asm volatile("global_load_async_to_lds_b128 %0, %1, %2"
                     :: "v"(lds + k * 2048u), "v"(voff + k * 2048u), "s"(Pb)
                     : "memory");
    }
}

// fp32 -> bf16 round-to-nearest-even
__global__ void k_convert(const float* __restrict__ P,
                          unsigned short* __restrict__ Pb, int n) {
    int i = blockIdx.x * blockDim.x + threadIdx.x;
    if (i < n) {
        unsigned u = __float_as_uint(P[i]);
        u += 0x7fffu + ((u >> 16) & 1u);
        Pb[i] = (unsigned short)(u >> 16);
    }
}

// Pass 1: per-row hard-negative max + has_pos/has_neg flags.
__global__ __launch_bounds__(128) void k_pass1(
    const unsigned short* __restrict__ Pb,
    const int* __restrict__ aff, const int* __restrict__ inst,
    float* __restrict__ hard, int* __restrict__ valid, int nrows)
{
    __shared__ unsigned short Btile[2][16 * DK];   // 2 x 8KB double buffer
    const int t     = threadIdx.x;
    const int lane  = t & 31;
    const int wave  = t >> 5;
    const int R     = (blockIdx.x * 4 + wave) * 16;
    const int nlane   = lane & 15;
    const int rowbase = R + ((lane >> 4) << 3);    // C/D: VGPR r -> row rowbase+r
    const unsigned lds0 = (unsigned)(uintptr_t)&Btile[0][0];
    const unsigned lds1 = (unsigned)(uintptr_t)&Btile[1][0];

    v16bf a[KCHUNKS];
    const int arow = R + nlane;
#pragma unroll
    for (int kc = 0; kc < KCHUNKS; ++kc) a[kc] = load_frag_g(Pb, arow, kc, lane);

    int aff_row[8], inst_row[8];
#pragma unroll
    for (int r = 0; r < 8; ++r) {
        aff_row[r]  = aff[rowbase + r];
        inst_row[r] = inst[rowbase + r];
    }

    float hmax[8];
#pragma unroll
    for (int r = 0; r < 8; ++r) hmax[r] = NEG_INF_F;
    int hasp = 0, hasn = 0;

    const int ntiles = nrows >> 4;
    async_issue_tile(Pb, 0, lds0, t);              // prologue: tile 0 -> buf0

    for (int ct = 0; ct < ntiles; ++ct) {
        if (ct + 1 < ntiles) {
            async_issue_tile(Pb, ct + 1, (ct & 1) ? lds0 : lds1, t);
            asm volatile("s_wait_asynccnt 0x4" ::: "memory");  // tile ct landed
        } else {
            asm volatile("s_wait_asynccnt 0x0" ::: "memory");
        }
        __syncthreads();                           // all waves' copies visible

        const unsigned short* tile = &Btile[ct & 1][0];
        v8f c = {};
#pragma unroll
        for (int kc = 0; kc < KCHUNKS; ++kc) {
            v16bf b = load_frag_lds(tile, kc, lane);
            c = __builtin_amdgcn_wmma_f32_16x16x32_bf16(
                    false, a[kc], false, b, (short)0, c, false, false);
        }
        const int col = (ct << 4) + nlane;
        const int ca  = aff[col];
        const int ci  = inst[col];
#pragma unroll
        for (int r = 0; r < 8; ++r) {              // branchless mask logic
            const bool neg = (aff_row[r] != ca);
            hmax[r] = fmaxf(hmax[r], neg ? c[r] : NEG_INF_F);
            hasn |= ((int)neg) << r;
            const bool pos = (!neg) & (inst_row[r] != ci);
            hasp |= ((int)pos) << r;
        }
        __syncthreads();                           // protect buffer reuse
    }

    // max / or across the 16 lanes of each half (xor 1..8 stays within a half)
#pragma unroll
    for (int r = 0; r < 8; ++r) {
#pragma unroll
        for (int s = 1; s < 16; s <<= 1)
            hmax[r] = fmaxf(hmax[r], __shfl_xor(hmax[r], s, 32));
    }
#pragma unroll
    for (int s = 1; s < 16; s <<= 1) {
        hasp |= __shfl_xor(hasp, s, 32);
        hasn |= __shfl_xor(hasn, s, 32);
    }

    if (nlane == 0) {
#pragma unroll
        for (int r = 0; r < 8; ++r) {
            hard[rowbase + r]  = hmax[r];
            valid[rowbase + r] = ((hasp >> r) & 1) & ((hasn >> r) & 1);
        }
    }
}

// Pass 2: margin losses over positive pairs of valid rows -> per-wave partials.
__global__ __launch_bounds__(128) void k_pass2(
    const unsigned short* __restrict__ Pb,
    const int* __restrict__ aff, const int* __restrict__ inst,
    const float* __restrict__ hard, const int* __restrict__ valid,
    float* __restrict__ ptotal, unsigned int* __restrict__ pcnt, int nrows)
{
    __shared__ unsigned short Btile[2][16 * DK];
    const int t     = threadIdx.x;
    const int lane  = t & 31;
    const int wave  = t >> 5;
    const int R     = (blockIdx.x * 4 + wave) * 16;
    const int nlane   = lane & 15;
    const int rowbase = R + ((lane >> 4) << 3);
    const unsigned lds0 = (unsigned)(uintptr_t)&Btile[0][0];
    const unsigned lds1 = (unsigned)(uintptr_t)&Btile[1][0];

    v16bf a[KCHUNKS];
    const int arow = R + nlane;
#pragma unroll
    for (int kc = 0; kc < KCHUNKS; ++kc) a[kc] = load_frag_g(Pb, arow, kc, lane);

    int aff_row[8], inst_row[8];
    float hr[8];
#pragma unroll
    for (int r = 0; r < 8; ++r) {
        const int i = rowbase + r;
        // invalid rows get a sentinel affordance -> never matches (ids are >=0)
        aff_row[r]  = valid[i] ? aff[i] : -1;
        inst_row[r] = inst[i];
        hr[r]       = hard[i];
    }

    float total = 0.0f;
    unsigned int cnt = 0;

    const int ntiles = nrows >> 4;
    async_issue_tile(Pb, 0, lds0, t);

    for (int ct = 0; ct < ntiles; ++ct) {
        if (ct + 1 < ntiles) {
            async_issue_tile(Pb, ct + 1, (ct & 1) ? lds0 : lds1, t);
            asm volatile("s_wait_asynccnt 0x4" ::: "memory");
        } else {
            asm volatile("s_wait_asynccnt 0x0" ::: "memory");
        }
        __syncthreads();

        const unsigned short* tile = &Btile[ct & 1][0];
        v8f c = {};
#pragma unroll
        for (int kc = 0; kc < KCHUNKS; ++kc) {
            v16bf b = load_frag_lds(tile, kc, lane);
            c = __builtin_amdgcn_wmma_f32_16x16x32_bf16(
                    false, a[kc], false, b, (short)0, c, false, false);
        }
        const int col = (ct << 4) + nlane;
        const int ca  = aff[col];
        const int ci  = inst[col];
#pragma unroll
        for (int r = 0; r < 8; ++r) {              // branchless
            const bool pos = (aff_row[r] == ca) & (inst_row[r] != ci);
            const float l  = fmaxf(hr[r] - c[r] + MARGIN_F, 0.0f);
            total += pos ? l : 0.0f;
            cnt   += (unsigned)pos;
        }
        __syncthreads();
    }

    // full-wave reduction (deterministic; no atomics)
#pragma unroll
    for (int s = 1; s < 32; s <<= 1) {
        total += __shfl_xor(total, s, 32);
        cnt   += __shfl_xor(cnt,   s, 32);
    }
    if (lane == 0) {
        const int wid = blockIdx.x * 4 + wave;
        ptotal[wid] = total;
        pcnt[wid]   = cnt;
    }
}

__global__ void k_final(const float* __restrict__ ptotal,
                        const unsigned int* __restrict__ pcnt,
                        int nw, float* __restrict__ out)
{
    if (threadIdx.x == 0 && blockIdx.x == 0) {
        double t = 0.0;
        unsigned long long c = 0;
        for (int i = 0; i < nw; ++i) { t += ptotal[i]; c += pcnt[i]; }
        out[0] = (c > 0) ? (float)(t / (double)c) : 0.0f;
    }
}

extern "C" void kernel_launch(void* const* d_in, const int* in_sizes, int n_in,
                              void* d_out, int out_size, void* d_ws, size_t ws_size,
                              hipStream_t stream) {
    const float* P    = (const float*)d_in[0];
    const int*   aff  = (const int*)d_in[1];
    const int*   inst = (const int*)d_in[2];
    float* out = (float*)d_out;

    const int nrows = in_sizes[1];        // 8192
    const int n     = nrows * DK;

    char* ws = (char*)d_ws;
    unsigned short* Pb = (unsigned short*)ws;
    size_t off = (size_t)n * sizeof(unsigned short);
    float* hard  = (float*)(ws + off);  off += (size_t)nrows * sizeof(float);
    int*   valid = (int*)(ws + off);    off += (size_t)nrows * sizeof(int);
    const int nwaves = nrows / 16;
    float*        ptotal = (float*)(ws + off);  off += (size_t)nwaves * sizeof(float);
    unsigned int* pcnt   = (unsigned int*)(ws + off);

    k_convert<<<(n + 255) / 256, 256, 0, stream>>>(P, Pb, n);

    dim3 grid(nrows / 64), blk(128);      // 4 waves/block, 16 rows/wave
    k_pass1<<<grid, blk, 0, stream>>>(Pb, aff, inst, hard, valid, nrows);
    k_pass2<<<grid, blk, 0, stream>>>(Pb, aff, inst, hard, valid, ptotal, pcnt, nrows);
    k_final<<<1, 32, 0, stream>>>(ptotal, pcnt, nwaves, out);
}
